// TransformerBlock_23871428231738
// MI455X (gfx1250) — matrix operand order repaired and checked by reading, not compiled, not run on hardware
//
#include <hip/hip_runtime.h>
#include <math.h>

// ---- CDNA5 WMMA vector types ----
typedef __attribute__((ext_vector_type(16))) __bf16 v16bf;
typedef __attribute__((ext_vector_type(8)))  float  v8f;

#define NH    12
#define BSZ   4
#define NSEQ  1024
#define DIM   768
#define C2    1536
#define CDIM  3072
#define MTOT  (BSZ*NSEQ)   // 4096

// =====================================================================
// Fused LN1 / LN2 of x -> bf16 activations (row = one (b,i), 768 cols)
// =====================================================================
__device__ inline float wave_sum32(float v) {
  #pragma unroll
  for (int o = 16; o > 0; o >>= 1) v += __shfl_xor(v, o, 32);
  return v;
}

__global__ __launch_bounds__(256)
void ln2x_kernel(const float* __restrict__ x,
                 const float* __restrict__ g1, const float* __restrict__ b1,
                 const float* __restrict__ g2, const float* __restrict__ b2,
                 __bf16* __restrict__ xa, __bf16* __restrict__ xm)
{
  const int row = blockIdx.x;
  const float* xr = x + (size_t)row * DIM;
  float v[3], s = 0.f, ss = 0.f;
  #pragma unroll
  for (int e = 0; e < 3; ++e) {
    v[e] = xr[threadIdx.x + e * 256];
    s += v[e]; ss += v[e] * v[e];
  }
  __shared__ float lsum[2][8];
  float sw = wave_sum32(s), ssw = wave_sum32(ss);
  const int w = threadIdx.x >> 5;
  if ((threadIdx.x & 31) == 0) { lsum[0][w] = sw; lsum[1][w] = ssw; }
  __syncthreads();
  __shared__ float mv[2];
  if (threadIdx.x == 0) {
    float S = 0.f, SS = 0.f;
    #pragma unroll
    for (int i = 0; i < 8; ++i) { S += lsum[0][i]; SS += lsum[1][i]; }
    float m = S * (1.0f / DIM);
    float var = SS * (1.0f / DIM) - m * m;
    mv[0] = m; mv[1] = rsqrtf(var + 1e-5f);
  }
  __syncthreads();
  const float m = mv[0], r = mv[1];
  #pragma unroll
  for (int e = 0; e < 3; ++e) {
    int c = threadIdx.x + e * 256;
    float t = (v[e] - m) * r;
    xa[(size_t)row * DIM + c] = (__bf16)(t * g1[c] + b1[c]);
    xm[(size_t)row * DIM + c] = (__bf16)(t * g2[c] + b2[c]);
  }
}

// =====================================================================
// LN over the gate half of h: h[row, 1536:3072] -> xg_ln[row, 0:1536]
// =====================================================================
__global__ __launch_bounds__(256)
void lng_kernel(const float* __restrict__ h,
                const float* __restrict__ g, const float* __restrict__ b,
                float* __restrict__ xg)
{
  const int row = blockIdx.x;
  const float* hr = h + (size_t)row * CDIM + C2;
  float v[6], s = 0.f, ss = 0.f;
  #pragma unroll
  for (int e = 0; e < 6; ++e) {
    v[e] = hr[threadIdx.x + e * 256];
    s += v[e]; ss += v[e] * v[e];
  }
  __shared__ float lsum[2][8];
  float sw = wave_sum32(s), ssw = wave_sum32(ss);
  const int w = threadIdx.x >> 5;
  if ((threadIdx.x & 31) == 0) { lsum[0][w] = sw; lsum[1][w] = ssw; }
  __syncthreads();
  __shared__ float mv[2];
  if (threadIdx.x == 0) {
    float S = 0.f, SS = 0.f;
    #pragma unroll
    for (int i = 0; i < 8; ++i) { S += lsum[0][i]; SS += lsum[1][i]; }
    float m = S * (1.0f / C2);
    float var = SS * (1.0f / C2) - m * m;
    mv[0] = m; mv[1] = rsqrtf(var + 1e-5f);
  }
  __syncthreads();
  const float m = mv[0], r = mv[1];
  #pragma unroll
  for (int e = 0; e < 6; ++e) {
    int c = threadIdx.x + e * 256;
    xg[(size_t)row * C2 + c] = (v[e] - m) * r * g[c] + b[c];
  }
}

// =====================================================================
// Weight convert + transpose to bf16: dst[n*K + k] = src[k*ld + off + n]
// =====================================================================
__global__ __launch_bounds__(256)
void cvt_t_kernel(const float* __restrict__ src, __bf16* __restrict__ dst,
                  int K, int N, int ld, int coloff)
{
  int idx = blockIdx.x * 256 + threadIdx.x;
  if (idx >= N * K) return;
  int n = idx / K, k = idx - n * K;
  dst[idx] = (__bf16)src[(size_t)k * ld + coloff + n];
}

// =====================================================================
// bf16 WMMA GEMM: out[m,n] = sum_k X[m,k]*Wt[n,k]  (+ epilogue)
// Wave tile 64x64 (16 accumulators), grid = (N/64, M/64), block = 1 wave.
// Per K-step: 16 b128 loads feed 16 WMMAs (1.0 loads/WMMA, ~32 FLOP/B L2).
// A frag per lane (l=lane&15, hb=lane>>4): row m0+l+16*mt,
//   K = hb*8 + {0..7} and 16 + hb*8 + {0..7} (interleaved per ISA A table).
// B frag per lane: col n0+16*nt+l, K = hb*8 + {0..7} and 16 + hb*8 + {0..7} (same as A).
// C: vgpr r -> row r + 8*hb, col = lane&15.
// =====================================================================
enum { EP_PLAIN = 0, EP_GELU = 1, EP_RESID = 2, EP_FINAL = 3 };

template<int EP>
__global__ __launch_bounds__(32)
void gemm_bf16_kernel(const __bf16* __restrict__ X, int lda,
                      const __bf16* __restrict__ Wt, int K,
                      const float* __restrict__ bias,
                      float* __restrict__ outf, __bf16* __restrict__ outb,
                      int ldout,
                      const float* __restrict__ xres, int ldx,
                      const float* __restrict__ scale)
{
  const int lane = threadIdx.x;
  const int l = lane & 15, hb = lane >> 4;
  const int n0 = blockIdx.x * 64;
  const int m0 = blockIdx.y * 64;

  const v8f vzero = {0.f, 0.f, 0.f, 0.f, 0.f, 0.f, 0.f, 0.f};
  v8f acc[4][4];
  #pragma unroll
  for (int i = 0; i < 4; ++i)
    #pragma unroll
    for (int j = 0; j < 4; ++j) acc[i][j] = vzero;

  const __bf16* abase = X + (size_t)(m0 + l) * lda + hb * 8;
  const __bf16* bbase = Wt + (size_t)(n0 + l) * K + hb * 8;

  for (int k0 = 0; k0 < K; k0 += 32) {
    union { uint4 u[2]; v16bf v; } fa[4], fb[4];
    #pragma unroll
    for (int mt = 0; mt < 4; ++mt) {
      const __bf16* ap = abase + (size_t)(mt * 16) * lda + k0;
      fa[mt].u[0] = *(const uint4*)(ap);
      fa[mt].u[1] = *(const uint4*)(ap + 16);
    }
    #pragma unroll
    for (int nt = 0; nt < 4; ++nt) {
      const __bf16* bp = bbase + (size_t)(nt * 16) * K + k0;
      fb[nt].u[0] = *(const uint4*)(bp);
      fb[nt].u[1] = *(const uint4*)(bp + 16);
    }
    #pragma unroll
    for (int mt = 0; mt < 4; ++mt)
      #pragma unroll
      for (int nt = 0; nt < 4; ++nt)
        acc[mt][nt] = __builtin_amdgcn_wmma_f32_16x16x32_bf16(
            false, fa[mt].v, false, fb[nt].v, (short)0, acc[mt][nt], false, false);
  }

  #pragma unroll
  for (int mt = 0; mt < 4; ++mt)
    #pragma unroll
    for (int nt = 0; nt < 4; ++nt) {
      const int col = n0 + nt * 16 + l;
      const float bc = bias[col];
      #pragma unroll
      for (int r = 0; r < 8; ++r) {
        const int row = m0 + mt * 16 + hb * 8 + r;
        float t = acc[mt][nt][r] + bc;
        if (EP == EP_PLAIN) {
          outf[(size_t)row * ldout + col] = t;
        } else if (EP == EP_GELU) {
          outf[(size_t)row * ldout + col] = 0.5f * t * (1.0f + erff(t * 0.70710678118654752f));
        } else if (EP == EP_RESID) {
          outb[(size_t)row * ldout + col] =
              (__bf16)(xres[(size_t)row * ldx + col] + scale[col] * t);
        } else { // EP_FINAL
          outf[(size_t)row * ldout + col] = t + xres[(size_t)row * ldx + col];
        }
      }
    }
}

// =====================================================================
// Depthwise conv (K=21, pad 10) on xg_ln along sequence, + conv bias,
// multiply by xr = h[:, :1536]  ->  u (bf16, row-major M x 1536)
// =====================================================================
__global__ __launch_bounds__(256)
void conv_mul_kernel(const float* __restrict__ xg,
                     const float* __restrict__ h,
                     const float* __restrict__ cw,
                     const float* __restrict__ cb,
                     __bf16* __restrict__ u)
{
  size_t idx = (size_t)blockIdx.x * 256 + threadIdx.x;
  if (idx >= (size_t)MTOT * C2) return;
  int ch  = (int)(idx % C2);
  int rowg = (int)(idx / C2);
  int b = rowg >> 10, i = rowg & 1023;
  float acc = cb[ch];
  #pragma unroll
  for (int k = 0; k < 21; ++k) {
    int ii = i + k - 10;
    if (ii >= 0 && ii < NSEQ)
      acc += xg[((size_t)(b * NSEQ + ii)) * C2 + ch] * cw[ch * 21 + k];
  }
  float xr = h[(size_t)rowg * CDIM + ch];
  u[idx] = (__bf16)(xr * acc);
}

// =====================================================================
// RoPE(1/sc) on first 32 dims of v per head, then sum over sequence.
// blocks: p*48 + b*12 + h ; 64 threads = j
// =====================================================================
__global__ __launch_bounds__(64)
void vsum_rope_kernel(const float* __restrict__ v0,
                      const float* __restrict__ v1,
                      float* __restrict__ vsum)
{
  const int blk = blockIdx.x;
  const int p = blk / 48, b = (blk / 12) & 3, hh = blk % 12;
  const float* V = (p == 0) ? v0 : v1;
  const int j = threadIdx.x;
  float acc = 0.f;
  const size_t base = ((size_t)b * NSEQ) * DIM + hh * 64;
  if (j < 32) {
    const int u   = j >> 1;
    const float inv  = powf(10000.f, -(float)(2 * u) * (1.0f / 32.0f));
    const int c   = j & 15;
    const float lbase = (2.0f * c + 12.8f) / 44.8f;
    const float sgn = (j & 1) ? 1.f : -1.f;
    for (int i = 0; i < NSEQ; ++i) {
      float ang = (float)i * inv;
      float cs = cosf(ang), sn = sinf(ang);
      float pw = (float)(i - 512) * (1.0f / 512.0f);
      float s = powf(lbase, -pw);           // 1/sc
      size_t off = base + (size_t)i * DIM;
      float x0 = V[off + j];
      float x1 = V[off + (j ^ 1)];
      acc += x0 * cs * s + sgn * x1 * sn * s;
    }
  } else {
    for (int i = 0; i < NSEQ; ++i) acc += V[base + (size_t)i * DIM + j];
  }
  vsum[(size_t)blk * 64 + j] = acc;
}

// =====================================================================
// attn per-batch vector: for each path, w = LN_12(colsum(R)); y = w⊗vsum;
// attn_vec[b] = sum_p (y_p[b] @ Wproj_p + bproj_p)
// =====================================================================
__global__ __launch_bounds__(256)
void attn_vec_kernel(const float* __restrict__ R,
                     const float* __restrict__ rg, const float* __restrict__ rb,
                     const float* __restrict__ Wproj,
                     const float* __restrict__ bproj,
                     const float* __restrict__ vsum,
                     float* __restrict__ attn_vec)
{
  const int b = blockIdx.x;
  __shared__ float y[DIM];
  __shared__ float w[NH];
  float acc[3] = {0.f, 0.f, 0.f};
  for (int p = 0; p < 2; ++p) {
    if (threadIdx.x == 0) {
      float cs[NH]; float S = 0.f;
      #pragma unroll
      for (int g = 0; g < NH; ++g) {
        float t = 0.f;
        for (int hh = 0; hh < NH; ++hh) t += R[p * NH * NH + hh * NH + g];
        cs[g] = t; S += t;
      }
      float m = S * (1.0f / NH);
      float var = 0.f;
      #pragma unroll
      for (int g = 0; g < NH; ++g) { float d = cs[g] - m; var += d * d; }
      var *= (1.0f / NH);
      float r = rsqrtf(var + 1e-5f);
      #pragma unroll
      for (int g = 0; g < NH; ++g)
        w[g] = (cs[g] - m) * r * rg[p * NH + g] + rb[p * NH + g];
    }
    __syncthreads();
    for (int c = threadIdx.x; c < DIM; c += 256)
      y[c] = w[c >> 6] * vsum[(((size_t)p * 4 + b) * NH + (c >> 6)) * 64 + (c & 63)];
    __syncthreads();
    #pragma unroll
    for (int e = 0; e < 3; ++e) {
      int o = threadIdx.x + e * 256;
      float a = bproj[p * DIM + o];
      for (int c = 0; c < DIM; ++c)
        a += y[c] * Wproj[((size_t)p * DIM + c) * DIM + o];
      acc[e] += a;
    }
    __syncthreads();
  }
  #pragma unroll
  for (int e = 0; e < 3; ++e)
    attn_vec[(size_t)b * DIM + threadIdx.x + e * 256] = acc[e];
}

// =====================================================================
// xf[:, 0:768] = bf16( x + ls1 * attn_vec[b] )
// =====================================================================
__global__ __launch_bounds__(256)
void xf_attn_kernel(const float* __restrict__ x,
                    const float* __restrict__ ls1,
                    const float* __restrict__ attn_vec,
                    __bf16* __restrict__ xf)
{
  size_t idx = (size_t)blockIdx.x * 256 + threadIdx.x;
  if (idx >= (size_t)MTOT * DIM) return;
  int c = (int)(idx % DIM);
  int row = (int)(idx / DIM);
  int b = row >> 10;
  float t = x[idx] + ls1[c] * attn_vec[(size_t)b * DIM + c];
  xf[(size_t)row * C2 + c] = (__bf16)t;
}

// =====================================================================
// Host-side launcher
// =====================================================================
extern "C" void kernel_launch(void* const* d_in, const int* in_sizes, int n_in,
                              void* d_out, int out_size, void* d_ws, size_t ws_size,
                              hipStream_t stream) {
  const float* x     = (const float*)d_in[0];
  const float* Wqkv  = (const float*)d_in[1];
  const float* bqkv  = (const float*)d_in[2];
  const float* Wproj = (const float*)d_in[3];
  const float* bproj = (const float*)d_in[4];
  const float* R     = (const float*)d_in[9];
  const float* rn_g  = (const float*)d_in[10];
  const float* rn_b  = (const float*)d_in[11];
  const float* ln1_g = (const float*)d_in[12];
  const float* ln1_b = (const float*)d_in[13];
  const float* ln2_g = (const float*)d_in[14];
  const float* ln2_b = (const float*)d_in[15];
  const float* ls1   = (const float*)d_in[16];
  const float* ls2   = (const float*)d_in[17];
  const float* W1    = (const float*)d_in[18];
  const float* b1    = (const float*)d_in[19];
  const float* gn_g  = (const float*)d_in[20];
  const float* gn_b  = (const float*)d_in[21];
  const float* cw    = (const float*)d_in[22];
  const float* cb    = (const float*)d_in[23];
  const float* W2    = (const float*)d_in[24];
  const float* b2    = (const float*)d_in[25];
  const float* Wout  = (const float*)d_in[26];
  const float* bout  = (const float*)d_in[27];
  float* out = (float*)d_out;

  // ---- workspace carve-up (aligned to 256B); peak ~125 MB ----
  char* base = (char*)d_ws;
  size_t off = 0;
  auto carve = [&](size_t bytes) -> char* {
    char* p = base + off;
    off += (bytes + 255) & ~(size_t)255;
    return p;
  };
  __bf16* xa    = (__bf16*)carve((size_t)MTOT * DIM * 2);
  __bf16* xm    = (__bf16*)carve((size_t)MTOT * DIM * 2);
  __bf16* Wv0t  = (__bf16*)carve((size_t)DIM * DIM * 2);
  __bf16* Wv1t  = (__bf16*)carve((size_t)DIM * DIM * 2);
  __bf16* W1t   = (__bf16*)carve((size_t)CDIM * DIM * 2);
  __bf16* W2t   = (__bf16*)carve((size_t)DIM * C2 * 2);
  __bf16* Woutt = (__bf16*)carve((size_t)DIM * C2 * 2);
  __bf16* xf    = (__bf16*)carve((size_t)MTOT * C2 * 2);
  float*  vsum  = (float*) carve((size_t)2 * BSZ * NH * 64 * 4);
  float*  avec  = (float*) carve((size_t)BSZ * DIM * 4);
  // big region: MLP phase uses h + xg + u; attention phase reuses it for v0/v1
  char* big = base + off;
  float*  h    = (float*)big;
  float*  xg   = (float*)(big + (((size_t)MTOT * CDIM * 4 + 255) & ~(size_t)255));
  __bf16* u    = (__bf16*)((char*)xg + (((size_t)MTOT * C2 * 4 + 255) & ~(size_t)255));
  float*  v0   = (float*)big;                               // after MLP GEMMs done
  float*  v1   = (float*)(big + (((size_t)MTOT * DIM * 4 + 255) & ~(size_t)255));

  // 1) fused LN1/LN2 -> bf16
  ln2x_kernel<<<MTOT, 256, 0, stream>>>(x, ln1_g, ln1_b, ln2_g, ln2_b, xa, xm);

  // 2) weight transpose-converts to bf16 [N x K] layouts
  {
    int n;
    n = DIM * DIM;
    cvt_t_kernel<<<(n + 255) / 256, 256, 0, stream>>>(Wqkv,                  Wv0t, DIM, DIM, 3 * DIM, 2 * DIM);
    cvt_t_kernel<<<(n + 255) / 256, 256, 0, stream>>>(Wqkv + (size_t)DIM * 3 * DIM, Wv1t, DIM, DIM, 3 * DIM, 2 * DIM);
    n = CDIM * DIM;
    cvt_t_kernel<<<(n + 255) / 256, 256, 0, stream>>>(W1,   W1t,   DIM, CDIM, CDIM, 0);
    n = DIM * C2;
    cvt_t_kernel<<<(n + 255) / 256, 256, 0, stream>>>(W2,   W2t,   C2,  DIM,  DIM,  0);
    cvt_t_kernel<<<(n + 255) / 256, 256, 0, stream>>>(Wout, Woutt, C2,  DIM,  DIM,  0);
  }

  // ---- MLP phase ----
  // 3) h = gelu(xm @ W1 + b1)
  gemm_bf16_kernel<EP_GELU><<<dim3(CDIM / 64, MTOT / 64), 32, 0, stream>>>(
      xm, DIM, W1t, DIM, b1, h, (__bf16*)nullptr, CDIM, nullptr, 0, nullptr);
  // 4) xg = LN(h[:, 1536:])
  lng_kernel<<<MTOT, 256, 0, stream>>>(h, gn_g, gn_b, xg);
  // 5) u = bf16( h[:, :1536] * (depthwise_conv(xg) + cb) )
  {
    size_t n = (size_t)MTOT * C2;
    conv_mul_kernel<<<(unsigned)((n + 255) / 256), 256, 0, stream>>>(xg, h, cw, cb, u);
  }
  // 6) xf[:, 768:] = bf16( x + ls2 * (u @ W2 + b2) )
  gemm_bf16_kernel<EP_RESID><<<dim3(DIM / 64, MTOT / 64), 32, 0, stream>>>(
      u, C2, W2t, C2, b2, nullptr, xf + DIM, C2, x, DIM, ls2);

  // ---- attention phase (big region now free: reuse for v0/v1) ----
  // 7) v_p = xa @ Wv_p + bqkv_p[1536:2304]
  gemm_bf16_kernel<EP_PLAIN><<<dim3(DIM / 64, MTOT / 64), 32, 0, stream>>>(
      xa, DIM, Wv0t, DIM, bqkv + 2 * DIM,              v0, (__bf16*)nullptr, DIM, nullptr, 0, nullptr);
  gemm_bf16_kernel<EP_PLAIN><<<dim3(DIM / 64, MTOT / 64), 32, 0, stream>>>(
      xa, DIM, Wv1t, DIM, bqkv + 3 * DIM + 2 * DIM,    v1, (__bf16*)nullptr, DIM, nullptr, 0, nullptr);
  // 8) vsum = sum_i rope(v, 1/sc)
  vsum_rope_kernel<<<2 * BSZ * NH, 64, 0, stream>>>(v0, v1, vsum);
  // 9) attn_vec[b] = sum_p (LN(colsum R_p) ⊗ vsum_p[b]) @ Wproj_p + bproj_p
  attn_vec_kernel<<<BSZ, 256, 0, stream>>>(R, rn_g, rn_b, Wproj, bproj, vsum, avec);
  // 10) xf[:, :768] = bf16( x + ls1 * attn_vec[b] )
  {
    size_t n = (size_t)MTOT * DIM;
    xf_attn_kernel<<<(unsigned)((n + 255) / 256), 256, 0, stream>>>(x, ls1, avec, xf);
  }
  // 11) out = xf @ Wout + bout + x
  gemm_bf16_kernel<EP_FINAL><<<dim3(DIM / 64, MTOT / 64), 32, 0, stream>>>(
      xf, C2, Woutt, C2, bout, out, (__bf16*)nullptr, DIM, x, DIM, nullptr);

  (void)in_sizes; (void)n_in; (void)out_size; (void)ws_size;
}
